// LinearAttention_45732811768070
// MI455X (gfx1250) — matrix operand order repaired and checked by reading, not compiled, not run on hardware
//
#include <hip/hip_runtime.h>

// Problem constants from the reference (setup_inputs is fixed).
#define BATCH 4
#define NSEQ  4096
#define NH    16
#define HD    64
#define HDIM  (NH * HD)      // 1024
#define SPLIT 16             // N-chunks per (b,h) in the KV pass
#define CHUNK (NSEQ / SPLIT) // 256

typedef __attribute__((ext_vector_type(16))) _Float16 v16h;
typedef __attribute__((ext_vector_type(4)))  _Float16 v4h;
typedef __attribute__((ext_vector_type(8)))  float    v8f;
typedef __attribute__((ext_vector_type(4)))  int      v4i;

// CDNA5 async global->LDS path (ASYNCcnt-tracked), if toolchain exposes it.
#if defined(__has_builtin)
#if __has_builtin(__builtin_amdgcn_global_load_async_to_lds_b128) && \
    __has_builtin(__builtin_amdgcn_s_wait_asynccnt)
#define HAVE_ASYNC_LDS 1
#endif
#endif
#ifndef HAVE_ASYNC_LDS
#define HAVE_ASYNC_LDS 0
#endif

#if HAVE_ASYNC_LDS
// 16B async copy: global (as1, v4i*) -> LDS (as3, v4i*), per-lane addresses.
// Builtin signature (from clang diagnostic): param0 is AS1 pointer to int4.
#define ASYNC_CP16(gp, lp)                                                     \
  __builtin_amdgcn_global_load_async_to_lds_b128(                              \
      (__attribute__((address_space(1))) v4i*)(uintptr_t)(gp),                 \
      (__attribute__((address_space(3))) v4i*)(uintptr_t)(lp), 0, 0)
#endif

// ---------------------------------------------------------------------------
// WMMA helpers (wave32, v_wmma_f32_16x16x32_f16)
// ---------------------------------------------------------------------------
static __device__ __forceinline__ v8f wmma16(v16h a, v16h b, v8f c) {
  return __builtin_amdgcn_wmma_f32_16x16x32_f16(false, a, false, b,
                                                (short)0, c, false, false);
}

// A fragment, 16x32 f16, from LDS where row dim == M and k runs contiguously.
static __device__ __forceinline__ v16h frag_a_lds(const _Float16* base,
                                                  int row0, int stride, int kofs) {
  const int lane = threadIdx.x & 31;
  const int laneM = lane & 15, hi = lane >> 4;
  const _Float16* p = base + (row0 + laneM) * stride + kofs;
  v16h a;
#pragma unroll
  for (int v = 0; v < 8; ++v) {
    const int k0 = 2 * v + hi * 16;
    a[2 * v]     = p[k0];
    a[2 * v + 1] = p[k0 + 1];
  }
  return a;
}

// B fragment, 32x16 f16, from LDS stored transposed: base[col][k] (k contiguous).
static __device__ __forceinline__ v16h frag_b_ldsT(const _Float16* base,
                                                   int col0, int stride, int kofs) {
  const int lane = threadIdx.x & 31;
  const int laneN = lane & 15, hi = lane >> 4;
  const _Float16* p = base + (col0 + laneN) * stride + kofs + hi * 16;
  v16h b;
#pragma unroll
  for (int v = 0; v < 8; ++v) {
    b[2 * v]     = p[2 * v];
    b[2 * v + 1] = p[2 * v + 1];
  }
  return b;
}

// B fragment, 32x16, gathered from global f32 stored as base[col][k] (k contig).
static __device__ __forceinline__ v16h frag_b_gf32(const float* base,
                                                   int col0, int stride, int kofs) {
  const int lane = threadIdx.x & 31;
  const int laneN = lane & 15, hi = lane >> 4;
  const float* p = base + (size_t)(col0 + laneN) * stride + kofs + hi * 16;
  v16h b;
#pragma unroll
  for (int v = 0; v < 8; ++v) {
    const float2 f = *reinterpret_cast<const float2*>(p + 2 * v);
    b[2 * v]     = (_Float16)f.x;
    b[2 * v + 1] = (_Float16)f.y;
  }
  return b;
}

static __device__ __forceinline__ float phi_f(float x) {
  return x > 0.f ? x + 1.f : __expf(x);  // elu(x)+1
}

// ---------------------------------------------------------------------------
// Kernel 0: zero the KV / Ksum accumulators in workspace
// ---------------------------------------------------------------------------
__global__ void la_zero_kernel(float* __restrict__ p, int n) {
  int i = blockIdx.x * 256 + threadIdx.x;
  if (i < n) p[i] = 0.f;
}

// ---------------------------------------------------------------------------
// Kernel 1: KV = phi(K)^T V (stored transposed [b][h][m][d]) and Ksum.
// grid = (SPLIT, H, B), 256 threads (8 waves), each wave owns 2 of 16 tiles.
// Streaming is double-buffered: async-to-LDS (CDNA5) when available, else
// register-pipelined float4 loads; either way next slab's loads overlap the
// current slab's WMMA work.
// ---------------------------------------------------------------------------
__global__ void __launch_bounds__(256)
la_kv_kernel(const float* __restrict__ keyp, const float* __restrict__ valp,
             float* __restrict__ wskv, float* __restrict__ wsks) {
  const int b = blockIdx.z, h = blockIdx.y, ck = blockIdx.x;
  const int tid = threadIdx.x;
  const int wave = tid >> 5;
  const int lane = tid & 31, laneN = lane & 15, hi = lane >> 4;

  __shared__ _Float16 kT[64 * 34];  // [d][n]  phi(K) transposed, padded stride
  __shared__ _Float16 vT[64 * 34];  // [m][n]  V transposed

  const int t0 = wave, t1 = wave + 8;
  const int tr0 = (t0 >> 2) * 16, tc0 = (t0 & 3) * 16;
  const int tr1 = (t1 >> 2) * 16, tc1 = (t1 & 3) * 16;

  v8f acc0 = {}; v8f acc1 = {};
  float ks = 0.f;
  const int n0 = ck * CHUNK;
  const float* kbase = keyp + (size_t)(b * NSEQ + n0) * HDIM + h * HD;
  const float* vbase = valp + (size_t)(b * NSEQ + n0) * HDIM + h * HD;

#if HAVE_ASYNC_LDS
  __shared__ float kraw[2][32 * 64];
  __shared__ float vraw[2][32 * 64];
  // Each thread issues 2 x b128 per matrix per 32x64 slab.
  const int e0 = tid * 4;            // first float of 1st b128 (within slab)
  const int e1 = (tid + 256) * 4;    // first float of 2nd b128
  const int an0 = e0 >> 6, ad0 = e0 & 63;
  const int an1 = e1 >> 6, ad1 = e1 & 63;

  {  // prologue: slab 0 into buffer 0
    ASYNC_CP16(kbase + (size_t)an0 * HDIM + ad0, &kraw[0][e0]);
    ASYNC_CP16(kbase + (size_t)an1 * HDIM + ad1, &kraw[0][e1]);
    ASYNC_CP16(vbase + (size_t)an0 * HDIM + ad0, &vraw[0][e0]);
    ASYNC_CP16(vbase + (size_t)an1 * HDIM + ad1, &vraw[0][e1]);
  }
  int buf = 0;
  const int f0 = tid * 8;            // convert: 8 consecutive floats per thread
  const int cn = f0 >> 6, cd = f0 & 63;

  for (int nb = 0; nb < CHUNK; nb += 32) {
    const bool has_next = (nb + 32) < CHUNK;
    if (has_next) {
      const float* kp = kbase + (size_t)(nb + 32) * HDIM;
      const float* vp = vbase + (size_t)(nb + 32) * HDIM;
      ASYNC_CP16(kp + (size_t)an0 * HDIM + ad0, &kraw[buf ^ 1][e0]);
      ASYNC_CP16(kp + (size_t)an1 * HDIM + ad1, &kraw[buf ^ 1][e1]);
      ASYNC_CP16(vp + (size_t)an0 * HDIM + ad0, &vraw[buf ^ 1][e0]);
      ASYNC_CP16(vp + (size_t)an1 * HDIM + ad1, &vraw[buf ^ 1][e1]);
      __builtin_amdgcn_s_wait_asynccnt(4);   // current slab done, next in flight
    } else {
      __builtin_amdgcn_s_wait_asynccnt(0);
    }
    __syncthreads();  // raw slab visible; previous wmma reads of kT/vT done
#pragma unroll
    for (int j = 0; j < 8; ++j) {
      kT[(cd + j) * 34 + cn] = (_Float16)phi_f(kraw[buf][f0 + j]);
      vT[(cd + j) * 34 + cn] = (_Float16)vraw[buf][f0 + j];
    }
    __syncthreads();

    if (tid < 64) {
#pragma unroll
      for (int r = 0; r < 32; ++r) ks += (float)kT[tid * 34 + r];
    }
    v16h a0 = frag_a_lds(kT, tr0, 34, 0);
    v16h b0 = frag_b_ldsT(vT, tc0, 34, 0);
    acc0 = wmma16(a0, b0, acc0);
    v16h a1 = frag_a_lds(kT, tr1, 34, 0);
    v16h b1 = frag_b_ldsT(vT, tc1, 34, 0);
    acc1 = wmma16(a1, b1, acc1);
    buf ^= 1;
  }
#else
  // Register double-buffer: 2 float4 per matrix per thread per slab.
  const int i40 = tid, i41 = tid + 256;          // float4 indices in 512
  const int rn0 = i40 >> 4, rd0 = (i40 & 15) * 4;
  const int rn1 = i41 >> 4, rd1 = (i41 & 15) * 4;
  float4 kr0, kr1, vr0, vr1;
  kr0 = *(const float4*)(kbase + (size_t)rn0 * HDIM + rd0);
  kr1 = *(const float4*)(kbase + (size_t)rn1 * HDIM + rd1);
  vr0 = *(const float4*)(vbase + (size_t)rn0 * HDIM + rd0);
  vr1 = *(const float4*)(vbase + (size_t)rn1 * HDIM + rd1);

  for (int nb = 0; nb < CHUNK; nb += 32) {
    __syncthreads();  // previous wmma reads of kT/vT done
    kT[(rd0 + 0) * 34 + rn0] = (_Float16)phi_f(kr0.x);
    kT[(rd0 + 1) * 34 + rn0] = (_Float16)phi_f(kr0.y);
    kT[(rd0 + 2) * 34 + rn0] = (_Float16)phi_f(kr0.z);
    kT[(rd0 + 3) * 34 + rn0] = (_Float16)phi_f(kr0.w);
    kT[(rd1 + 0) * 34 + rn1] = (_Float16)phi_f(kr1.x);
    kT[(rd1 + 1) * 34 + rn1] = (_Float16)phi_f(kr1.y);
    kT[(rd1 + 2) * 34 + rn1] = (_Float16)phi_f(kr1.z);
    kT[(rd1 + 3) * 34 + rn1] = (_Float16)phi_f(kr1.w);
    vT[(rd0 + 0) * 34 + rn0] = (_Float16)vr0.x;
    vT[(rd0 + 1) * 34 + rn0] = (_Float16)vr0.y;
    vT[(rd0 + 2) * 34 + rn0] = (_Float16)vr0.z;
    vT[(rd0 + 3) * 34 + rn0] = (_Float16)vr0.w;
    vT[(rd1 + 0) * 34 + rn1] = (_Float16)vr1.x;
    vT[(rd1 + 1) * 34 + rn1] = (_Float16)vr1.y;
    vT[(rd1 + 2) * 34 + rn1] = (_Float16)vr1.z;
    vT[(rd1 + 3) * 34 + rn1] = (_Float16)vr1.w;
    if (nb + 32 < CHUNK) {  // issue next slab now; waits land next iteration
      const float* kp = kbase + (size_t)(nb + 32) * HDIM;
      const float* vp = vbase + (size_t)(nb + 32) * HDIM;
      kr0 = *(const float4*)(kp + (size_t)rn0 * HDIM + rd0);
      kr1 = *(const float4*)(kp + (size_t)rn1 * HDIM + rd1);
      vr0 = *(const float4*)(vp + (size_t)rn0 * HDIM + rd0);
      vr1 = *(const float4*)(vp + (size_t)rn1 * HDIM + rd1);
    }
    __syncthreads();

    if (tid < 64) {
#pragma unroll
      for (int r = 0; r < 32; ++r) ks += (float)kT[tid * 34 + r];
    }
    v16h a0 = frag_a_lds(kT, tr0, 34, 0);
    v16h b0 = frag_b_ldsT(vT, tc0, 34, 0);
    acc0 = wmma16(a0, b0, acc0);
    v16h a1 = frag_a_lds(kT, tr1, 34, 0);
    v16h b1 = frag_b_ldsT(vT, tc1, 34, 0);
    acc1 = wmma16(a1, b1, acc1);
  }
#endif

  // Accumulate partial KV into workspace, stored transposed: wskv[bh][m][d].
  float* kvb = wskv + (size_t)(b * NH + h) * (HD * HD);
#pragma unroll
  for (int v = 0; v < 8; ++v) {
    atomicAdd(&kvb[(tc0 + laneN) * HD + tr0 + hi * 8 + v], acc0[v]);
    atomicAdd(&kvb[(tc1 + laneN) * HD + tr1 + hi * 8 + v], acc1[v]);
  }
  if (tid < 64) atomicAdd(&wsks[(b * NH + h) * HD + tid], ks);
}

// ---------------------------------------------------------------------------
// Kernel 2: Y = Z * (phi(Q) @ KV) per head, then out = Y @ W_out^T + b_out.
// grid = (N/16, B), 256 threads. Waves 0..7: 2 heads each; waves 0..3: proj.
// ---------------------------------------------------------------------------
__global__ void __launch_bounds__(256)
la_out_kernel(const float* __restrict__ qp, const float* __restrict__ wskv,
              const float* __restrict__ wsks, const float* __restrict__ wout,
              const float* __restrict__ bout, float* __restrict__ out) {
  const int b = blockIdx.y;
  const int n0 = blockIdx.x * 16;
  const int tid = threadIdx.x;
  const int wave = tid >> 5;
  const int lane = tid & 31, laneN = lane & 15, hi = lane >> 4;

  __shared__ _Float16 qs[16 * 1032];  // phi(Q), 16 rows x 1024 (+8 pad)
  __shared__ _Float16 ys[16 * 1032];  // Y, 16 rows x 1024 (+8 pad)
  __shared__ float zs[NH][16];        // Z[h][row]
  __shared__ float kss[HDIM];         // Ksum for all heads of this batch

  // Stage Ksum (coalesced float4).
  {
    const float4 f = *(const float4*)(wsks + (size_t)b * HDIM + tid * 4);
    *(float4*)(&kss[tid * 4]) = f;
  }

  // Stage phi(Q): the 16x1024 block is contiguous (4096 float4s). Batch loads
  // 4-at-a-time into registers so they overlap, then convert + packed store.
  const float* qbase = qp + (size_t)(b * NSEQ + n0) * HDIM;
#pragma unroll
  for (int g = 0; g < 4; ++g) {
    float4 t[4];
#pragma unroll
    for (int j = 0; j < 4; ++j)
      t[j] = *(const float4*)(qbase + (size_t)(tid + (g * 4 + j) * 256) * 4);
#pragma unroll
    for (int j = 0; j < 4; ++j) {
      const int e = (tid + (g * 4 + j) * 256) * 4;
      const int r = e >> 10, c = e & 1023;
      v4h hq;
      hq[0] = (_Float16)phi_f(t[j].x);
      hq[1] = (_Float16)phi_f(t[j].y);
      hq[2] = (_Float16)phi_f(t[j].z);
      hq[3] = (_Float16)phi_f(t[j].w);
      *(v4h*)(&qs[r * 1032 + c]) = hq;
    }
  }
  __syncthreads();

  // Z[h][r] = 1 / (phi(Q)[r,h,:] . Ksum[h] + eps)
  {
    const int h = tid >> 4, r = tid & 15;
    float dot = 0.f;
#pragma unroll
    for (int d = 0; d < HD; ++d)
      dot += (float)qs[r * 1032 + h * HD + d] * kss[h * HD + d];
    zs[h][r] = 1.f / (dot + 1e-6f);
  }
  __syncthreads();

  // Y_h[16,64] = phi(Q_h)[16,64] @ KV_h[64,64], rows scaled by Z.
#pragma unroll
  for (int hh = 0; hh < 2; ++hh) {
    const int h = wave * 2 + hh;
    const float* kvb = wskv + (size_t)(b * NH + h) * (HD * HD);  // [m][d]
#pragma unroll
    for (int ct = 0; ct < 4; ++ct) {
      v8f acc = {};
#pragma unroll
      for (int kb = 0; kb < 2; ++kb) {
        v16h a  = frag_a_lds(qs, 0, 1032, h * HD + kb * 32);
        v16h bb = frag_b_gf32(kvb, ct * 16, HD, kb * 32);
        acc = wmma16(a, bb, acc);
      }
#pragma unroll
      for (int v = 0; v < 8; ++v) {
        const int m = v + hi * 8;
        ys[m * 1032 + h * HD + ct * 16 + laneN] = (_Float16)(acc[v] * zs[h][m]);
      }
    }
  }
  __syncthreads();

  // out[16,64] = Y[16,1024] @ W_out^T[1024,64] + b_out. W_out is [64][1024]
  // row-major, so B[k][j] = wout[j*1024 + k] with k contiguous.
  if (wave < 4) {
    const int j0 = wave * 16;
    v8f acc = {};
#pragma unroll 2
    for (int kb = 0; kb < 32; ++kb) {
      v16h a  = frag_a_lds(ys, 0, 1032, kb * 32);
      v16h bb = frag_b_gf32(wout, j0, HDIM, kb * 32);
      acc = wmma16(a, bb, acc);
    }
    const float bo = bout[j0 + laneN];
#pragma unroll
    for (int v = 0; v < 8; ++v) {
      const int m = v + hi * 8;
      out[((size_t)(b * NSEQ + n0 + m)) * HD + j0 + laneN] = acc[v] + bo;
    }
  }
}

// ---------------------------------------------------------------------------
extern "C" void kernel_launch(void* const* d_in, const int* in_sizes, int n_in,
                              void* d_out, int out_size, void* d_ws, size_t ws_size,
                              hipStream_t stream) {
  const float* query = (const float*)d_in[0];
  const float* key   = (const float*)d_in[1];
  const float* value = (const float*)d_in[2];
  const float* wout  = (const float*)d_in[3];
  const float* bout  = (const float*)d_in[4];
  float* out = (float*)d_out;

  float* wskv = (float*)d_ws;                        // B*H*64*64 f32, [b][h][m][d]
  float* wsks = wskv + (size_t)BATCH * NH * HD * HD; // B*H*64 f32
  const int zn = BATCH * NH * HD * HD + BATCH * NH * HD;

  la_zero_kernel<<<(zn + 255) / 256, 256, 0, stream>>>(wskv, zn);

  dim3 g1(SPLIT, NH, BATCH);
  la_kv_kernel<<<g1, 256, 0, stream>>>(key, value, wskv, wsks);

  dim3 g2(NSEQ / 16, BATCH);
  la_out_kernel<<<g2, 256, 0, stream>>>(query, wskv, wsks, wout, bout, out);
}
